// EDDG_4870492913707
// MI455X (gfx1250) — matrix operand order repaired and checked by reading, not compiled, not run on hardware
//
#include <hip/hip_runtime.h>
#include <hip/hip_bf16.h>

// ---------------------------------------------------------------------------
// MI455X (gfx1250, wave32) implementation.
//   - All 1x1-conv layers run on v_wmma_f32_16x16x32_bf16 (fp32 accum).
//   - BN+ReLU fused as per-channel (scale, shift) applied while staging the
//     next GEMM's input tile into LDS (already arranged in WMMA B-fragment
//     layout so each lane reads its fragment with two b128 LDS loads).
//   - Inner loop: 8 ds_load_b128 issued as one batch into 4 distinct fragment
//     register sets, single dscnt wait, then 4 back-to-back WMMAs.
//   - kNN (K=16) done with per-lane register top-16 over LDS-staged chunks,
//     covariance + closed-form 3x3 eigvalsh + tiny MLP fused in one kernel.
// ---------------------------------------------------------------------------

#define BB    2
#define NPTS  8192
#define KNN   16

typedef __attribute__((ext_vector_type(16))) __bf16 v16bf;
typedef __attribute__((ext_vector_type(8)))  float  v8f;

__device__ __forceinline__ unsigned short f2bf(float f) {
    unsigned int u = __float_as_uint(f);
    unsigned int r = u + 0x7FFFu + ((u >> 16) & 1u);   // round-to-nearest-even
    return (unsigned short)(r >> 16);
}

// ---------------------------------------------------------------------------
// Convert fp32 weights (M x Kin) -> bf16 bits (M x Kpad), zero padded.
// ---------------------------------------------------------------------------
__global__ void wconv_kernel(const float* __restrict__ W,
                             unsigned short* __restrict__ Wb,
                             int M, int Kin, int Kpad) {
    int i = blockIdx.x * blockDim.x + threadIdx.x;
    if (i >= M * Kpad) return;
    int m = i / Kpad, k = i - m * Kpad;
    float v = (k < Kin) ? W[m * Kin + k] : 0.0f;
    Wb[i] = f2bf(v);
}

// ---------------------------------------------------------------------------
// Y[b,m,n] = sum_k Wb[m,k] * act(X[b,k,n]) + bias[m]
//   act(x) = aff ? relu(aff[k].x * x + aff[k].y) : x
// Block: 256 threads = 8 waves; wave w computes rows m0 = by*128 + w*16,
// columns n0..n0+63 (4 WMMA accumulators of 16 cols each).  K stepped by 32.
// ---------------------------------------------------------------------------
__global__ __launch_bounds__(256) void gemm_bn_kernel(
    const float* __restrict__ X,            // (B, K, N) fp32
    const unsigned short* __restrict__ Wb,  // (M, Kpad) bf16 bits, Kpad % 32 == 0
    const float* __restrict__ bias,         // (M)
    const float2* __restrict__ aff,         // (K) scale/shift (+relu) or null
    float* __restrict__ Y,                  // (B, M, N)
    int M, int Kpad, int N)
{
    // LDS B-tile held directly in WMMA B-fragment layout:
    //   [subtile s(0..3)][lane(0..31)][elem e(0..15)]  (bf16)
    __shared__ alignas(32) unsigned short ldsB[4 * 32 * 16];

    const int tid    = threadIdx.x;
    const int wave   = tid >> 5;
    const int lane   = tid & 31;
    const int lsub   = lane & 15;
    const int laneHi = lane >> 4;            // 0 or 1
    const int n0     = blockIdx.x * 64;
    const int m0     = blockIdx.y * 128 + wave * 16;
    const int b      = blockIdx.z;
    const float* Xb  = X + (size_t)b * Kpad * N;
    const bool active = (m0 < M);

    v8f acc[4] = {};

    for (int kk = 0; kk < Kpad; kk += 32) {
        // ---- A fragment from global weights: issue before the barrier so the
        //      load latency overlaps the LDS staging of the X tile.
        // ISA A-layout (16x32, 16-bit): lane<16: e0..7 -> K=kk+0..7,
        // e8..15 -> K=kk+16..23; lane>=16 shifted by +8.
        union { uint4 q[2]; v16bf v; } afrag;
        if (active) {
            const unsigned short* wrow =
                Wb + (size_t)(m0 + lsub) * Kpad + kk + laneHi * 8;
            afrag.q[0] = *(const uint4*)(wrow);        // 8 halves
            afrag.q[1] = *(const uint4*)(wrow + 16);   // 8 halves
        }

        __syncthreads();
        // ---- stage 32(K) x 64(N) fp32 tile -> bf16 B-fragments in LDS ----
        #pragma unroll
        for (int it = 0; it < 8; ++it) {
            int lin = it * 256 + tid;        // 0..2047
            int k   = lin >> 6;              // 0..31
            int n   = lin & 63;              // 0..63 (coalesced along n)
            int ch  = kk + k;
            float v = Xb[(size_t)ch * N + n0 + n];
            if (aff) { float2 st = aff[ch]; v = fmaxf(st.x * v + st.y, 0.0f); }
            // B-layout matching A: lanes 0-15 hold K=0..7 (e0..7), K=16..23 (e8..15);
            // lanes 16-31 hold K=8..15, K=24..31; column = lane%16.
            int s  = n >> 4;
            int fl = (n & 15) + (((k >> 3) & 1) << 4);
            int fe = (k & 7) + ((k >> 4) << 3);
            ldsB[(s * 32 + fl) * 16 + fe] = f2bf(v);
        }
        __syncthreads();

        // prefetch next K-tile of X into caches (global_prefetch_b8)
        if (kk + 32 < Kpad) {
            int pr = tid >> 3, pc = (tid & 7) * 8;
            __builtin_prefetch(&Xb[(size_t)(kk + 32 + pr) * N + n0 + pc], 0, 3);
        }

        if (active) {
            // ---- load all 4 B fragments first (one DS clause, one wait),
            //      then issue the 4 WMMAs back-to-back (independent accs).
            union { uint4 q[2]; v16bf v; } bfrag[4];
            #pragma unroll
            for (int s = 0; s < 4; ++s) {
                const unsigned short* bp = &ldsB[(s * 32 + lane) * 16];
                bfrag[s].q[0] = *(const uint4*)(bp);
                bfrag[s].q[1] = *(const uint4*)(bp + 8);
            }
            #pragma unroll
            for (int s = 0; s < 4; ++s) {
                acc[s] = __builtin_amdgcn_wmma_f32_16x16x32_bf16(
                    false, afrag.v, false, bfrag[s].v, (short)0, acc[s],
                    false, false);
            }
        }
    }

    if (active) {
        float* Yb = Y + (size_t)b * M * N;
        #pragma unroll
        for (int s = 0; s < 4; ++s) {
            int col = n0 + s * 16 + lsub;
            #pragma unroll
            for (int r = 0; r < 8; ++r) {
                int m = m0 + r + laneHi * 8;   // C/D layout: VGPR r -> row r / 8+r
                Yb[(size_t)m * N + col] = acc[s][r] + bias[m];
            }
        }
    }
}

// ---------------------------------------------------------------------------
// Per-channel BN stats over (B, N): one block per channel, writes fused
// (scale, shift):  y' = relu(s*y + t)  with s = g*rsqrt(var+eps), t = be - m*s
// ---------------------------------------------------------------------------
__global__ __launch_bounds__(256) void bn_stats_kernel(
    const float* __restrict__ Y, const float* __restrict__ g,
    const float* __restrict__ be, float2* __restrict__ aff, int M, int N)
{
    __shared__ float s_sum[256], s_sq[256];
    int c = blockIdx.x;
    float sum = 0.f, sq = 0.f;
    for (int b = 0; b < BB; ++b) {
        const float* p = Y + ((size_t)b * M + c) * N;
        for (int n = threadIdx.x; n < N; n += 256) {
            float v = p[n]; sum += v; sq += v * v;
        }
    }
    s_sum[threadIdx.x] = sum; s_sq[threadIdx.x] = sq;
    __syncthreads();
    for (int off = 128; off > 0; off >>= 1) {
        if (threadIdx.x < off) {
            s_sum[threadIdx.x] += s_sum[threadIdx.x + off];
            s_sq[threadIdx.x]  += s_sq[threadIdx.x + off];
        }
        __syncthreads();
    }
    if (threadIdx.x == 0) {
        float inv = 1.0f / (float)(BB * N);
        float m   = s_sum[0] * inv;
        float var = s_sq[0] * inv - m * m;
        float s   = g[c] * rsqrtf(var + 1e-5f);
        aff[c]    = make_float2(s, be[c] - m * s);
    }
}

// ---------------------------------------------------------------------------
// kNN (K=16) + covariance + 3x3 eigvalsh (ascending) + 3->4->4 MLP.
// One query point per lane; candidates streamed through LDS as float4(x,y,z,|p|^2).
// ---------------------------------------------------------------------------
__global__ __launch_bounds__(256) void knn_h3_kernel(
    const float* __restrict__ pc,   // (B, N, 3)
    const float* __restrict__ w1, const float* __restrict__ b1,  // (3,4),(4)
    const float* __restrict__ w2, const float* __restrict__ b2,  // (4,4),(4)
    float* __restrict__ h3)         // (B, 4, N)
{
    __shared__ float4 pts[256];
    const int q = blockIdx.x * 256 + threadIdx.x;
    const int b = blockIdx.y;
    const float* P = pc + (size_t)b * NPTS * 3;

    float qx = P[q * 3], qy = P[q * 3 + 1], qz = P[q * 3 + 2];
    float qsq = qx * qx + qy * qy + qz * qz;

    float bd[KNN]; int bi[KNN];
    #pragma unroll
    for (int j = 0; j < KNN; ++j) { bd[j] = 3.4e38f; bi[j] = 0; }
    float worst = 3.4e38f;

    for (int c0 = 0; c0 < NPTS; c0 += 256) {
        __syncthreads();
        {
            int i = c0 + threadIdx.x;
            float x = P[i * 3], y = P[i * 3 + 1], z = P[i * 3 + 2];
            pts[threadIdx.x] = make_float4(x, y, z, x * x + y * y + z * z);
        }
        __syncthreads();
        for (int j = 0; j < 256; ++j) {
            float4 p = pts[j];
            float d = qsq + p.w - 2.0f * (qx * p.x + qy * p.y + qz * p.z);
            if (d < worst) {                      // rare path (~K ln(N/K) times)
                int idx = c0 + j;
                bool done = false;
                #pragma unroll
                for (int t = 0; t < KNN; ++t) {   // replace first slot == worst
                    bool rep = (!done) && (bd[t] == worst);
                    if (rep) { bd[t] = d; bi[t] = idx; done = true; }
                }
                worst = bd[0];
                #pragma unroll
                for (int t = 1; t < KNN; ++t) worst = fmaxf(worst, bd[t]);
            }
        }
    }

    // gather neighbors (xyz is L2-resident: 96KB/batch), mean + covariance
    float nx[KNN], ny[KNN], nz[KNN];
    float mx = 0.f, my = 0.f, mz = 0.f;
    #pragma unroll
    for (int j = 0; j < KNN; ++j) {
        const float* pp = P + (size_t)bi[j] * 3;
        nx[j] = pp[0]; ny[j] = pp[1]; nz[j] = pp[2];
        mx += nx[j]; my += ny[j]; mz += nz[j];
    }
    const float invK = 1.0f / (float)KNN;
    mx *= invK; my *= invK; mz *= invK;
    float cxx=0,cyy=0,czz=0,cxy=0,cxz=0,cyz=0;
    #pragma unroll
    for (int j = 0; j < KNN; ++j) {
        float dx = nx[j]-mx, dy = ny[j]-my, dz = nz[j]-mz;
        cxx += dx*dx; cyy += dy*dy; czz += dz*dz;
        cxy += dx*dy; cxz += dx*dz; cyz += dy*dz;
    }
    cxx*=invK; cyy*=invK; czz*=invK; cxy*=invK; cxz*=invK; cyz*=invK;

    // closed-form symmetric 3x3 eigenvalues (ascending e0<=e1<=e2)
    float e0, e1, e2;
    {
        float p1 = cxy*cxy + cxz*cxz + cyz*cyz;
        float qm = (cxx + cyy + czz) * (1.0f/3.0f);
        float aa = cxx - qm, bb = cyy - qm, cc = czz - qm;
        float p2 = aa*aa + bb*bb + cc*cc + 2.0f*p1;
        float p  = sqrtf(p2 * (1.0f/6.0f));
        if (p < 1e-20f) { e0 = e1 = e2 = qm; }
        else {
            float ip  = 1.0f / p;
            float A00 = aa*ip, A11 = bb*ip, A22 = cc*ip;
            float A01 = cxy*ip, A02 = cxz*ip, A12 = cyz*ip;
            float detB = A00*(A11*A22 - A12*A12)
                       - A01*(A01*A22 - A12*A02)
                       + A02*(A01*A12 - A11*A02);
            float r = fminf(1.0f, fmaxf(-1.0f, 0.5f * detB));
            float phi = acosf(r) * (1.0f/3.0f);
            float big = qm + 2.0f*p*cosf(phi);
            float sml = qm + 2.0f*p*cosf(phi + 2.0943951023931953f);
            e0 = sml; e1 = 3.0f*qm - big - sml; e2 = big;
        }
    }

    // h3 = relu(ev @ w1 + b1) @ w2 + b2
    float t[4];
    #pragma unroll
    for (int k = 0; k < 4; ++k)
        t[k] = fmaxf(e0*w1[k] + e1*w1[4+k] + e2*w1[8+k] + b1[k], 0.0f);
    float* H = h3 + (size_t)b * 4 * NPTS;
    #pragma unroll
    for (int o = 0; o < 4; ++o) {
        float v = b2[o];
        #pragma unroll
        for (int k = 0; k < 4; ++k) v += t[k] * w2[k*4 + o];
        H[(size_t)o * NPTS + q] = v;
    }
}

// ---------------------------------------------------------------------------
// z (B, 192, N) fp32, zero-padded 164->192:
//   ch 0..127  = h1 raw
//   ch 128..159= relu(aff3 applied to Y3)
//   ch 160..163= h3
// ---------------------------------------------------------------------------
__global__ void zassm_kernel(const float* __restrict__ h1,
                             const float* __restrict__ Y3,
                             const float2* __restrict__ aff3,
                             const float* __restrict__ h3,
                             float* __restrict__ z)
{
    size_t i = (size_t)blockIdx.x * blockDim.x + threadIdx.x;
    size_t total = (size_t)BB * 192 * NPTS;
    if (i >= total) return;
    int n = (int)(i % NPTS);
    int c = (int)((i / NPTS) % 192);
    int b = (int)(i / ((size_t)192 * NPTS));
    float v;
    if (c < 128) {
        v = h1[((size_t)b * 128 + c) * NPTS + n];
    } else if (c < 160) {
        int cc = c - 128; float2 st = aff3[cc];
        v = fmaxf(st.x * Y3[((size_t)b * 32 + cc) * NPTS + n] + st.y, 0.0f);
    } else if (c < 164) {
        v = h3[((size_t)b * 4 + (c - 160)) * NPTS + n];
    } else {
        v = 0.0f;
    }
    z[i] = v;
}

// ---------------------------------------------------------------------------
// Output: [xyz (B,N,3)] ++ [relu(bn(Yc3)) (B,128,N)]
// ---------------------------------------------------------------------------
__global__ void finalize_kernel(const float* __restrict__ pc,
                                const float* __restrict__ Yc3,
                                const float2* __restrict__ affC3,
                                float* __restrict__ out)
{
    size_t i = (size_t)blockIdx.x * blockDim.x + threadIdx.x;
    const size_t nxyz = (size_t)BB * NPTS * 3;
    const size_t nz   = (size_t)BB * 128 * NPTS;
    if (i < nxyz) {
        out[i] = pc[i];
    } else if (i < nxyz + nz) {
        size_t j = i - nxyz;
        int c = (int)((j / NPTS) % 128);
        float2 st = affC3[c];
        out[i] = fmaxf(st.x * Yc3[j] + st.y, 0.0f);
    }
}

// ---------------------------------------------------------------------------
extern "C" void kernel_launch(void* const* d_in, const int* in_sizes, int n_in,
                              void* d_out, int out_size, void* d_ws, size_t ws_size,
                              hipStream_t stream) {
    (void)in_sizes; (void)n_in; (void)out_size; (void)ws_size;
    const float* pc    = (const float*)d_in[0];
    const float* h1    = (const float*)d_in[1];
    const float* h2    = (const float*)d_in[2];
    const float* DG_w1 = (const float*)d_in[4];  const float* DG_b1 = (const float*)d_in[5];
    const float* DG_g1 = (const float*)d_in[6];  const float* DG_be1= (const float*)d_in[7];
    const float* DG_w2 = (const float*)d_in[8];  const float* DG_b2 = (const float*)d_in[9];
    const float* DG_g2 = (const float*)d_in[10]; const float* DG_be2= (const float*)d_in[11];
    const float* DG_w3 = (const float*)d_in[12]; const float* DG_b3 = (const float*)d_in[13];
    const float* DG_g3 = (const float*)d_in[14]; const float* DG_be3= (const float*)d_in[15];
    const float* ED_w1 = (const float*)d_in[16]; const float* ED_b1 = (const float*)d_in[17];
    const float* ED_w2 = (const float*)d_in[18]; const float* ED_b2 = (const float*)d_in[19];
    const float* C_w1  = (const float*)d_in[20]; const float* C_b1  = (const float*)d_in[21];
    const float* C_g1  = (const float*)d_in[22]; const float* C_be1 = (const float*)d_in[23];
    const float* C_w2  = (const float*)d_in[24]; const float* C_b2  = (const float*)d_in[25];
    const float* C_g2  = (const float*)d_in[26]; const float* C_be2 = (const float*)d_in[27];
    const float* C_w3  = (const float*)d_in[28]; const float* C_b3  = (const float*)d_in[29];
    const float* C_g3  = (const float*)d_in[30]; const float* C_be3 = (const float*)d_in[31];

    char* base = (char*)d_ws; size_t off = 0;
    auto alloc = [&](size_t bytes) -> void* {
        void* p = base + off; off += (bytes + 255) & ~(size_t)255; return p;
    };
    // bf16 weight buffers (Kpad multiples of 32)
    unsigned short* W1b = (unsigned short*)alloc((size_t)256*1024*2);
    unsigned short* W2b = (unsigned short*)alloc((size_t)64*256*2);
    unsigned short* W3b = (unsigned short*)alloc((size_t)32*64*2);
    unsigned short* Cw1b= (unsigned short*)alloc((size_t)512*192*2);
    unsigned short* Cw2b= (unsigned short*)alloc((size_t)256*512*2);
    unsigned short* Cw3b= (unsigned short*)alloc((size_t)128*256*2);
    // fused BN scale/shift
    float2* aff1  = (float2*)alloc(256*sizeof(float2));
    float2* aff2  = (float2*)alloc(64*sizeof(float2));
    float2* aff3  = (float2*)alloc(32*sizeof(float2));
    float2* affC1 = (float2*)alloc(512*sizeof(float2));
    float2* affC2 = (float2*)alloc(256*sizeof(float2));
    float2* affC3 = (float2*)alloc(128*sizeof(float2));
    // activations (raw conv outputs)
    float* Y1   = (float*)alloc((size_t)BB*256*NPTS*4);
    float* Y2   = (float*)alloc((size_t)BB*64*NPTS*4);
    float* Y3   = (float*)alloc((size_t)BB*32*NPTS*4);
    float* h3b  = (float*)alloc((size_t)BB*4*NPTS*4);
    float* zbuf = (float*)alloc((size_t)BB*192*NPTS*4);
    float* Yc1  = (float*)alloc((size_t)BB*512*NPTS*4);
    float* Yc2  = (float*)alloc((size_t)BB*256*NPTS*4);
    float* Yc3  = (float*)alloc((size_t)BB*128*NPTS*4);

    auto cdiv = [](size_t a, size_t b) { return (unsigned)((a + b - 1) / b); };

    // weight conversion
    wconv_kernel<<<cdiv(256*1024,256),256,0,stream>>>(DG_w1, W1b, 256,1024,1024);
    wconv_kernel<<<cdiv(64*256,  256),256,0,stream>>>(DG_w2, W2b, 64, 256, 256);
    wconv_kernel<<<cdiv(32*64,   256),256,0,stream>>>(DG_w3, W3b, 32, 64,  64);
    wconv_kernel<<<cdiv(512*192, 256),256,0,stream>>>(C_w1,  Cw1b,512,164, 192);
    wconv_kernel<<<cdiv(256*512, 256),256,0,stream>>>(C_w2,  Cw2b,256,512, 512);
    wconv_kernel<<<cdiv(128*256, 256),256,0,stream>>>(C_w3,  Cw3b,128,256, 256);

    const dim3 blk(256);
    // DG chain (BN+ReLU of layer i fused into layer i+1's input staging)
    gemm_bn_kernel<<<dim3(NPTS/64, 2, BB), blk, 0, stream>>>(h2, W1b, DG_b1, nullptr, Y1, 256, 1024, NPTS);
    bn_stats_kernel<<<256, blk, 0, stream>>>(Y1, DG_g1, DG_be1, aff1, 256, NPTS);
    gemm_bn_kernel<<<dim3(NPTS/64, 1, BB), blk, 0, stream>>>(Y1, W2b, DG_b2, aff1, Y2, 64, 256, NPTS);
    bn_stats_kernel<<<64, blk, 0, stream>>>(Y2, DG_g2, DG_be2, aff2, 64, NPTS);
    gemm_bn_kernel<<<dim3(NPTS/64, 1, BB), blk, 0, stream>>>(Y2, W3b, DG_b3, aff2, Y3, 32, 64, NPTS);
    bn_stats_kernel<<<32, blk, 0, stream>>>(Y3, DG_g3, DG_be3, aff3, 32, NPTS);

    // geometric branch: kNN -> covariance -> eigvalsh -> tiny MLP
    knn_h3_kernel<<<dim3(NPTS/256, BB), blk, 0, stream>>>(pc, ED_w1, ED_b1, ED_w2, ED_b2, h3b);

    // concat into padded 192-channel buffer
    zassm_kernel<<<cdiv((size_t)BB*192*NPTS,256), blk, 0, stream>>>(h1, Y3, aff3, h3b, zbuf);

    // classifier chain
    gemm_bn_kernel<<<dim3(NPTS/64, 4, BB), blk, 0, stream>>>(zbuf, Cw1b, C_b1, nullptr, Yc1, 512, 192, NPTS);
    bn_stats_kernel<<<512, blk, 0, stream>>>(Yc1, C_g1, C_be1, affC1, 512, NPTS);
    gemm_bn_kernel<<<dim3(NPTS/64, 2, BB), blk, 0, stream>>>(Yc1, Cw2b, C_b2, affC1, Yc2, 256, 512, NPTS);
    bn_stats_kernel<<<256, blk, 0, stream>>>(Yc2, C_g2, C_be2, affC2, 256, NPTS);
    gemm_bn_kernel<<<dim3(NPTS/64, 1, BB), blk, 0, stream>>>(Yc2, Cw3b, C_b3, affC2, Yc3, 128, 256, NPTS);
    bn_stats_kernel<<<128, blk, 0, stream>>>(Yc3, C_g3, C_be3, affC3, 128, NPTS);

    // outputs: xyz copy + final BN+ReLU apply
    size_t total_out = (size_t)BB*NPTS*3 + (size_t)BB*128*NPTS;
    finalize_kernel<<<cdiv(total_out,256), blk, 0, stream>>>(pc, Yc3, affC3, (float*)d_out);
}
